// CausalGATLayer_74483322847926
// MI455X (gfx1250) — matrix operand order repaired and checked by reading, not compiled, not run on hardware
//
#include <hip/hip_runtime.h>
#include <hip/hip_bf16.h>
#include <math.h>

#define N 512
#define IN 256
#define HID 256
#define OUT 256
#define HD 64
#define KPAD 416   // 385 padded to multiple of 32

typedef __attribute__((ext_vector_type(16))) __bf16 v16bf;
typedef __attribute__((ext_vector_type(8)))  float  v8f;

// ---------------------------------------------------------------------------
// WMMA GEMM: C[M x Nc] = act(X[M x K] @ op(W) + bias)
//   TRANSB=1: W is (Nc x K) row-major (use W^T);  TRANSB=0: W is (K x Nc).
// One wave per 16x16 tile; K advances by 32 per v_wmma_f32_16x16x32_bf16.
// A fragment (contiguous k order chosen here; AMD's table order differs): lane half h, elem e ->
//   K = 16h + e  => one contiguous 16-float run.
// B fragment (same contiguous k order as A): K = 16h + e => one 16-float run (^T).
// ---------------------------------------------------------------------------
template <int TRANSB>
__global__ __launch_bounds__(32)
void wmma_gemm(const float* __restrict__ X, int ldx,
               const float* __restrict__ W, int ldw,
               const float* __restrict__ bias, int relu,
               float* __restrict__ C, int ldc, int K) {
  const int lane = threadIdx.x & 31;
  const int half = lane >> 4;
  const int mn   = lane & 15;
  const int tn   = blockIdx.x * 16;
  const int tm   = blockIdx.y * 16;

  const float* xrow = X + (tm + mn) * ldx + 16 * half;  // A row, half-offset
  const float* wb;
  if (TRANSB) wb = W + (tn + mn) * ldw + 16 * half;     // B^T row, half-offset
  else        wb = W + (tn + mn) + 16 * half * ldw;     // B col, half-offset

  v8f c = {0.f, 0.f, 0.f, 0.f, 0.f, 0.f, 0.f, 0.f};

  for (int kb = 0; kb < K; kb += 32) {
    // ---- A fragment: 4x float4 (one contiguous 16-float run) ----
    float4 a0 = *(const float4*)(xrow + kb);
    float4 a1 = *(const float4*)(xrow + kb + 4);
    float4 a2 = *(const float4*)(xrow + kb + 8);
    float4 a3 = *(const float4*)(xrow + kb + 12);
    v16bf a;
    a[0] = (__bf16)a0.x; a[1] = (__bf16)a0.y; a[2] = (__bf16)a0.z; a[3] = (__bf16)a0.w;
    a[4] = (__bf16)a1.x; a[5] = (__bf16)a1.y; a[6] = (__bf16)a1.z; a[7] = (__bf16)a1.w;
    a[8]  = (__bf16)a2.x; a[9]  = (__bf16)a2.y; a[10] = (__bf16)a2.z; a[11] = (__bf16)a2.w;
    a[12] = (__bf16)a3.x; a[13] = (__bf16)a3.y; a[14] = (__bf16)a3.z; a[15] = (__bf16)a3.w;

    v16bf b;
    if (TRANSB) {
      // ---- B^T fragment: one contiguous 16-float run ----
      float4 b0 = *(const float4*)(wb + kb);
      float4 b1 = *(const float4*)(wb + kb + 4);
      float4 b2 = *(const float4*)(wb + kb + 8);
      float4 b3 = *(const float4*)(wb + kb + 12);
      b[0] = (__bf16)b0.x; b[1] = (__bf16)b0.y; b[2] = (__bf16)b0.z; b[3] = (__bf16)b0.w;
      b[4] = (__bf16)b1.x; b[5] = (__bf16)b1.y; b[6] = (__bf16)b1.z; b[7] = (__bf16)b1.w;
      b[8]  = (__bf16)b2.x; b[9]  = (__bf16)b2.y; b[10] = (__bf16)b2.z; b[11] = (__bf16)b2.w;
      b[12] = (__bf16)b3.x; b[13] = (__bf16)b3.y; b[14] = (__bf16)b3.z; b[15] = (__bf16)b3.w;
    } else {
      // ---- B fragment: 16 strided loads down the K dimension ----
      const float* p = wb + kb * ldw;
#pragma unroll
      for (int e = 0; e < 16; ++e) b[e] = (__bf16)p[e * ldw];
    }

    c = __builtin_amdgcn_wmma_f32_16x16x32_bf16(false, a, false, b,
                                                (short)0, c, false, false);
  }

  // Branch-free epilogue: single bias load (col is invariant over r),
  // relu folded into a fmax floor (-inf when relu disabled).
  const int col = tn + mn;
  const float bv    = bias ? bias[col] : 0.f;
  const float floor = relu ? 0.f : -__builtin_inff();
  float* crow = C + tm * ldc + col;
#pragma unroll
  for (int r = 0; r < 8; ++r) {
    crow[(r + 8 * half) * ldc] = fmaxf(c[r] + bv, floor);
  }
}

// ---------------------------------------------------------------------------
// u = Wh1 @ sa0, v = Wh1 @ sa1   (one block per row)
// ---------------------------------------------------------------------------
__global__ void matvec_uv(const float* __restrict__ Wh1,
                          const float* __restrict__ sa0,
                          const float* __restrict__ sa1,
                          float* __restrict__ u, float* __restrict__ v) {
  __shared__ float r0[256], r1[256];
  int i = blockIdx.x, t = threadIdx.x;
  float x = Wh1[i * HID + t];
  r0[t] = x * sa0[t];
  r1[t] = x * sa1[t];
  __syncthreads();
  for (int s = 128; s > 0; s >>= 1) {
    if (t < s) { r0[t] += r0[t + s]; r1[t] += r1[t + s]; }
    __syncthreads();
  }
  if (t == 0) { u[i] = r0[0]; v[i] = r1[0]; }
}

// ---------------------------------------------------------------------------
// Row-wise masked softmax: P[i,j] = softmax_j( adj ? u_i+v_j : -inf )
// ---------------------------------------------------------------------------
__global__ void attn_softmax(const float* __restrict__ u,
                             const float* __restrict__ v,
                             const int* __restrict__ adj,
                             float* __restrict__ P) {
  __shared__ float red[256];
  __shared__ float mx_s, z_s;
  const float NEG = -__builtin_inff();
  int i = blockIdx.x, t = threadIdx.x;
  float ui = u[i];
  float v0 = adj[i * N + t]       ? ui + v[t]       : NEG;
  float v1 = adj[i * N + t + 256] ? ui + v[t + 256] : NEG;
  red[t] = fmaxf(v0, v1);
  __syncthreads();
  for (int s = 128; s > 0; s >>= 1) {
    if (t < s) red[t] = fmaxf(red[t], red[t + s]);
    __syncthreads();
  }
  if (t == 0) mx_s = red[0];
  __syncthreads();
  float mx = mx_s;
  float e0 = expf(v0 - mx), e1 = expf(v1 - mx);
  red[t] = e0 + e1;
  __syncthreads();
  for (int s = 128; s > 0; s >>= 1) {
    if (t < s) red[t] += red[t + s];
    __syncthreads();
  }
  if (t == 0) z_s = red[0];
  __syncthreads();
  float iz = 1.f / z_s;
  P[i * N + t]       = e0 * iz;
  P[i * N + t + 256] = e1 * iz;
}

// g[k] = sum_d ce_w2[d,k]*(ca0+ca1)[d];  scal[2] = ce_b2 . (ca0+ca1)
__global__ void make_g(const float* __restrict__ ce_w2,
                       const float* __restrict__ ce_b2,
                       const float* __restrict__ ca0,
                       const float* __restrict__ ca1,
                       float* __restrict__ g, float* __restrict__ scal) {
  int t = threadIdx.x;
  float acc = 0.f;
  for (int d = 0; d < HD; ++d) acc += ce_w2[d * HID + t] * (ca0[d] + ca1[d]);
  g[t] = acc;
  if (t == 0) {
    float c0 = 0.f;
    for (int d = 0; d < HD; ++d) c0 += ce_b2[d] * (ca0[d] + ca1[d]);
    scal[2] = c0;
  }
}

// s2[i,j] = relu(A_i + B_j + b) . g + c0, diag = -inf
__global__ void s2_kernel(const float* __restrict__ Aw,
                          const float* __restrict__ Bw,
                          const float* __restrict__ ce_b1,
                          const float* __restrict__ g,
                          const float* __restrict__ scal,
                          float* __restrict__ s2) {
  __shared__ float sA[256], sB[256], sG[256];
  int i = blockIdx.x, t = threadIdx.x;
  sA[t] = Aw[i * HID + t];
  sB[t] = ce_b1[t];
  sG[t] = g[t];
  __syncthreads();
  float c0 = scal[2];
  const float NEG = -__builtin_inff();
  for (int jj = 0; jj < 2; ++jj) {
    int j = t + 256 * jj;
    const float* Bj = Bw + j * HID;
    float acc = 0.f;
    for (int k = 0; k < HID; ++k)
      acc += fmaxf(sA[k] + Bj[k] + sB[k], 0.f) * sG[k];
    s2[i * N + j] = (j == i) ? NEG : acc + c0;
  }
}

__global__ void row_max(const float* __restrict__ s2, float* __restrict__ rmax) {
  __shared__ float red[256];
  int i = blockIdx.x, t = threadIdx.x;
  red[t] = fmaxf(s2[i * N + t], s2[i * N + t + 256]);
  __syncthreads();
  for (int s = 128; s > 0; s >>= 1) {
    if (t < s) red[t] = fmaxf(red[t], red[t + s]);
    __syncthreads();
  }
  if (t == 0) rmax[i] = red[0];
}

__global__ void final_max(const float* __restrict__ rmax,
                          float* __restrict__ scal, float* __restrict__ Sacc) {
  __shared__ float red[256];
  int t = threadIdx.x;
  Sacc[t] = 0.f;  // zero accumulator for pass C
  red[t] = fmaxf(rmax[t], rmax[t + 256]);
  __syncthreads();
  for (int s = 128; s > 0; s >>= 1) {
    if (t < s) red[t] = fmaxf(red[t], red[t + s]);
    __syncthreads();
  }
  if (t == 0) scal[0] = red[0];
}

__global__ void row_sumexp(const float* __restrict__ s2,
                           const float* __restrict__ scal,
                           float* __restrict__ rsum) {
  __shared__ float red[256];
  int i = blockIdx.x, t = threadIdx.x;
  float mx = scal[0];
  red[t] = expf(s2[i * N + t] - mx) + expf(s2[i * N + t + 256] - mx);
  __syncthreads();
  for (int s = 128; s > 0; s >>= 1) {
    if (t < s) red[t] += red[t + s];
    __syncthreads();
  }
  if (t == 0) rsum[i] = red[0];
}

__global__ void final_sum(const float* __restrict__ rsum, float* __restrict__ scal) {
  __shared__ float red[256];
  int t = threadIdx.x;
  red[t] = rsum[t] + rsum[t + 256];
  __syncthreads();
  for (int s = 128; s > 0; s >>= 1) {
    if (t < s) red[t] += red[t + s];
    __syncthreads();
  }
  if (t == 0) scal[1] = red[0];
}

// Sacc[k] += sum_j exp(s2[i,j]-mx) * relu(A[i,k]+B[j,k]+b[k])
__global__ void passC(const float* __restrict__ Aw, const float* __restrict__ Bw,
                      const float* __restrict__ ce_b1,
                      const float* __restrict__ s2,
                      const float* __restrict__ scal, float* __restrict__ Sacc) {
  __shared__ float wsh[256];
  int i = blockIdx.x, t = threadIdx.x;     // t == k
  float mx = scal[0];
  float base = Aw[i * HID + t] + ce_b1[t];
  float acc = 0.f;
  for (int jb = 0; jb < N; jb += 256) {
    __syncthreads();
    wsh[t] = expf(s2[i * N + jb + t] - mx);
    __syncthreads();
    for (int jj = 0; jj < 256; ++jj)
      acc += wsh[jj] * fmaxf(base + Bw[(jb + jj) * HID + t], 0.f);
  }
  atomicAdd(&Sacc[t], acc);
}

// H2row[d] = (Sacc . ce_w2[d,:]) / Z + ce_b2[d]
__global__ void h2_row(const float* __restrict__ Sacc,
                       const float* __restrict__ ce_w2,
                       const float* __restrict__ ce_b2,
                       const float* __restrict__ scal, float* __restrict__ H2row) {
  int d = threadIdx.x;
  float iz = 1.f / scal[1];
  float acc = 0.f;
  for (int k = 0; k < HID; ++k) acc += Sacc[k] * ce_w2[d * HID + k];
  H2row[d] = acc * iz + ce_b2[d];
}

__global__ void concat2(const float* __restrict__ V, const float* __restrict__ ph,
                        float* __restrict__ X) {
  int i = blockIdx.x, t = threadIdx.x;
  X[i * 512 + t]       = V[i * IN + t];
  X[i * 512 + 256 + t] = ph[i * HID + t];
}

// H3[i,:] = prefix-mean of tf rows (causal softmax of row-constant scores)
__global__ void h3_prefix(const float* __restrict__ tf, float* __restrict__ H3) {
  int d = threadIdx.x;  // 64 threads
  float run = 0.f;
  for (int i = 0; i < N; ++i) {
    run += tf[i * HD + d];
    H3[i * HD + d] = run / (float)(i + 1);
  }
}

// Find first two neighbors, build Xcat3 = [V, n0, n1]
__global__ void h4_prep(const int* __restrict__ adj, const float* __restrict__ V,
                        float* __restrict__ X) {
  __shared__ int si0, si1;
  int i = blockIdx.x, t = threadIdx.x;
  if (t == 0) {
    int i0 = -1, i1 = -1;
    for (int j = 0; j < N; ++j) {
      if (adj[i * N + j]) {
        if (i0 < 0) i0 = j;
        else { i1 = j; break; }
      }
    }
    si0 = i0; si1 = i1;
  }
  __syncthreads();
  int i0 = si0, i1 = si1;
  bool valid = (i1 >= 0);
  float* row = X + i * 768;
  row[t]       = V[i * IN + t];
  row[256 + t] = valid ? V[i0 * IN + t] : 0.f;
  row[512 + t] = valid ? V[i1 * IN + t] : 0.f;
}

__global__ void h4_sum(const float* __restrict__ cf, float* __restrict__ H4v) {
  int d = threadIdx.x;  // 64 threads
  float acc = 0.f;
  for (int i = 0; i < N; ++i) acc += cf[i * HD + d];
  H4v[d] = acc;
}

// Hcp (512 x 416): [H1 | H2row | H3 | H4col | zero-pad]
__global__ void build_hcp(const float* __restrict__ H1, const float* __restrict__ H2row,
                          const float* __restrict__ H3, const float* __restrict__ H4v,
                          float* __restrict__ Hcp) {
  int i = blockIdx.x, t = threadIdx.x;  // 416 threads
  float val;
  if (t < 256)      val = H1[i * HID + t];
  else if (t < 320) val = H2row[t - 256];
  else if (t < 384) val = H3[i * HD + (t - 320)];
  else if (t == 384) val = (i < HD) ? H4v[i] : 0.f;
  else              val = 0.f;
  Hcp[i * KPAD + t] = val;
}

__global__ void build_w2p(const float* __restrict__ W2, float* __restrict__ W2p) {
  int r = blockIdx.x, t = threadIdx.x;  // 416 threads
  W2p[r * KPAD + t] = (t < 385) ? W2[r * 385 + t] : 0.f;
}

// LayerNorm + ELU
__global__ void ln_elu(const float* __restrict__ X, const float* __restrict__ g,
                       const float* __restrict__ b, float* __restrict__ out) {
  __shared__ float red[256];
  __shared__ float mu_s, var_s;
  int i = blockIdx.x, t = threadIdx.x;
  float x = X[i * OUT + t];
  red[t] = x;
  __syncthreads();
  for (int s = 128; s > 0; s >>= 1) {
    if (t < s) red[t] += red[t + s];
    __syncthreads();
  }
  if (t == 0) mu_s = red[0] * (1.f / 256.f);
  __syncthreads();
  float d = x - mu_s;
  red[t] = d * d;
  __syncthreads();
  for (int s = 128; s > 0; s >>= 1) {
    if (t < s) red[t] += red[t + s];
    __syncthreads();
  }
  if (t == 0) var_s = red[0] * (1.f / 256.f);
  __syncthreads();
  float y = d * rsqrtf(var_s + 1e-5f) * g[t] + b[t];
  out[i * OUT + t] = (y > 0.f) ? y : (expf(y) - 1.f);
}

// ---------------------------------------------------------------------------
extern "C" void kernel_launch(void* const* d_in, const int* in_sizes, int n_in,
                              void* d_out, int out_size, void* d_ws, size_t ws_size,
                              hipStream_t stream) {
  const float* V     = (const float*)d_in[0];
  const int*   adj   = (const int*)  d_in[1];
  const float* prevh = (const float*)d_in[2];
  const float* W1    = (const float*)d_in[3];
  const float* sa0   = (const float*)d_in[4];
  const float* sa1   = (const float*)d_in[5];
  const float* ce_w1 = (const float*)d_in[6];
  const float* ce_b1 = (const float*)d_in[7];
  const float* ce_w2 = (const float*)d_in[8];
  const float* ce_b2 = (const float*)d_in[9];
  const float* ca0   = (const float*)d_in[10];
  const float* ca1   = (const float*)d_in[11];
  const float* te_w1 = (const float*)d_in[12];
  const float* te_b1 = (const float*)d_in[13];
  const float* te_w2 = (const float*)d_in[14];
  const float* te_b2 = (const float*)d_in[15];
  const float* pe_w1 = (const float*)d_in[18];
  const float* pe_b1 = (const float*)d_in[19];
  const float* pe_w2 = (const float*)d_in[20];
  const float* pe_b2 = (const float*)d_in[21];
  const float* W2    = (const float*)d_in[24];
  const float* op_w  = (const float*)d_in[25];
  const float* op_b  = (const float*)d_in[26];
  const float* ln_g  = (const float*)d_in[27];
  const float* ln_b  = (const float*)d_in[28];
  float* out = (float*)d_out;

  float* w = (float*)d_ws;
  size_t o = 0;
  auto alloc = [&](size_t n) { float* p = w + o; o += n; return p; };
  float* Wh1   = alloc(N * HID);
  float* Aw    = alloc(N * HID);
  float* Bw    = alloc(N * HID);
  float* P     = alloc(N * N);      // reused as s2 after H1 GEMM
  float* H1    = alloc(N * HID);
  float* u     = alloc(N);
  float* vv    = alloc(N);
  float* rmax  = alloc(N);
  float* rsum  = alloc(N);
  float* scal  = alloc(8);          // [0]=mx [1]=Z [2]=c0
  float* g     = alloc(HID);
  float* Sacc  = alloc(HID);
  float* H2row = alloc(HD);
  float* Xcat2 = alloc(N * 512);
  float* t1    = alloc(N * HID);    // reused as c1
  float* tf    = alloc(N * HD);
  float* H3    = alloc(N * HD);
  float* Xcat3 = alloc(N * 768);
  float* cf    = alloc(N * HD);
  float* H4v   = alloc(HD);
  float* Hcp   = alloc(N * KPAD);
  float* W2p   = alloc(OUT * KPAD);
  float* Hc2   = alloc(N * OUT);
  float* out2  = alloc(N * OUT);
  (void)ws_size; (void)n_in; (void)in_sizes; (void)out_size;

  dim3 wv(32);
  // --- H1 path: Wh1 = V @ W1^T ; masked row softmax ; H1 = P @ Wh1
  wmma_gemm<1><<<dim3(HID / 16, N / 16), wv, 0, stream>>>(V, IN, W1, IN, nullptr, 0, Wh1, HID, IN);
  matvec_uv<<<N, 256, 0, stream>>>(Wh1, sa0, sa1, u, vv);
  attn_softmax<<<N, 256, 0, stream>>>(u, vv, adj, P);
  wmma_gemm<0><<<dim3(HID / 16, N / 16), wv, 0, stream>>>(P, N, Wh1, HID, nullptr, 0, H1, HID, N);

  // --- H2 path (factored pairwise term)
  wmma_gemm<1><<<dim3(HID / 16, N / 16), wv, 0, stream>>>(V, IN, ce_w1, 2 * IN, nullptr, 0, Aw, HID, IN);
  wmma_gemm<1><<<dim3(HID / 16, N / 16), wv, 0, stream>>>(V, IN, ce_w1 + IN, 2 * IN, nullptr, 0, Bw, HID, IN);
  make_g<<<1, 256, 0, stream>>>(ce_w2, ce_b2, ca0, ca1, g, scal);
  float* s2 = P;  // reuse: P consumed by H1 GEMM above
  s2_kernel<<<N, 256, 0, stream>>>(Aw, Bw, ce_b1, g, scal, s2);
  row_max<<<N, 256, 0, stream>>>(s2, rmax);
  final_max<<<1, 256, 0, stream>>>(rmax, scal, Sacc);
  row_sumexp<<<N, 256, 0, stream>>>(s2, scal, rsum);
  final_sum<<<1, 256, 0, stream>>>(rsum, scal);
  passC<<<N, 256, 0, stream>>>(Aw, Bw, ce_b1, s2, scal, Sacc);
  h2_row<<<1, 64, 0, stream>>>(Sacc, ce_w2, ce_b2, scal, H2row);

  // --- H3 path: tf = MLP([V, prev]); H3 = prefix mean
  concat2<<<N, 256, 0, stream>>>(V, prevh, Xcat2);
  wmma_gemm<1><<<dim3(HID / 16, N / 16), wv, 0, stream>>>(Xcat2, 512, te_w1, 512, te_b1, 1, t1, HID, 512);
  wmma_gemm<1><<<dim3(HD / 16, N / 16), wv, 0, stream>>>(t1, HID, te_w2, HID, te_b2, 0, tf, HD, HID);
  h3_prefix<<<1, 64, 0, stream>>>(tf, H3);

  // --- H4 path: first-two-neighbor MLP, column-summed
  h4_prep<<<N, 256, 0, stream>>>(adj, V, Xcat3);
  float* c1 = t1;  // reuse
  wmma_gemm<1><<<dim3(HID / 16, N / 16), wv, 0, stream>>>(Xcat3, 768, pe_w1, 768, pe_b1, 1, c1, HID, 768);
  wmma_gemm<1><<<dim3(HD / 16, N / 16), wv, 0, stream>>>(c1, HID, pe_w2, HID, pe_b2, 0, cf, HD, HID);
  h4_sum<<<1, 64, 0, stream>>>(cf, H4v);

  // --- Combine, project, layernorm, elu
  build_hcp<<<N, KPAD, 0, stream>>>(H1, H2row, H3, H4v, Hcp);
  build_w2p<<<OUT, KPAD, 0, stream>>>(W2, W2p);
  wmma_gemm<1><<<dim3(OUT / 16, N / 16), wv, 0, stream>>>(Hcp, KPAD, W2p, KPAD, nullptr, 0, Hc2, OUT, KPAD);
  wmma_gemm<1><<<dim3(OUT / 16, N / 16), wv, 0, stream>>>(Hc2, OUT, op_w, OUT, op_b, 0, out2, OUT, OUT);
  ln_elu<<<N, 256, 0, stream>>>(out2, ln_g, ln_b, out);
}
